// RoutedGQAAttention_63660005262009
// MI455X (gfx1250) — compile-verified, matrix operand order checked by reading, not run on hardware
//
#include <hip/hip_runtime.h>

// Routed GQA attention for gfx1250 (MI455X): all GEMMs + attention via
// v_wmma_f32_16x16x32_bf16 (f32 in -> packed bf16 operands, f32 accumulate).

#define HIDDIM 2048
#define NHEADS 32
#define NKVH   8
#define HDIM   64
#define NEXP   4
#define NTOK   2048

typedef __attribute__((ext_vector_type(16))) __bf16    v16bf;
typedef __attribute__((ext_vector_type(8)))  float     v8f;
typedef __attribute__((ext_vector_type(8)))  unsigned  v8u;

__device__ __forceinline__ unsigned rne_hi(float f) {
  unsigned u = __float_as_uint(f);
  u += 0x7FFFu + ((u >> 16) & 1u);            // round-to-nearest-even
  return u >> 16;
}

__device__ __forceinline__ __bf16 f2bf(float f) {
  unsigned short h = (unsigned short)rne_hi(f);
  return __builtin_bit_cast(__bf16, h);
}

// Packed f32x2 -> bf16x2 (one VALU op on gfx1250 if the builtin exists).
__device__ __forceinline__ unsigned pk2bf(float lo, float hi) {
#if __has_builtin(__builtin_amdgcn_cvt_pk_bf16_f32)
  auto p = __builtin_amdgcn_cvt_pk_bf16_f32(lo, hi);
  return __builtin_bit_cast(unsigned, p);
#else
  return rne_hi(lo) | (rne_hi(hi) << 16);
#endif
}

// Build a 16-element bf16 WMMA fragment from two contiguous 8-float runs.
__device__ __forceinline__ v16bf mk_frag(const float* __restrict__ p, float scale) {
  v8u f;
#pragma unroll
  for (int q = 0; q < 4; ++q) {
    f[q]     = pk2bf(p[2 * q] * scale,      p[2 * q + 1] * scale);
    f[4 + q] = pk2bf(p[16 + 2 * q] * scale, p[16 + 2 * q + 1] * scale);
  }
  return __builtin_bit_cast(v16bf, f);
}

// Same, but elements strided by `stride` floats (column gather).
__device__ __forceinline__ v16bf mk_frag_strided(const float* __restrict__ p, size_t stride) {
  v8u f;
#pragma unroll
  for (int q = 0; q < 4; ++q) {
    f[q]     = pk2bf(p[(size_t)(2 * q) * stride],      p[(size_t)(2 * q + 1) * stride]);
    f[4 + q] = pk2bf(p[(size_t)(16 + 2 * q) * stride], p[(size_t)(17 + 2 * q) * stride]);
  }
  return __builtin_bit_cast(v16bf, f);
}

__device__ __forceinline__ v8f wmma_bf16(v16bf a, v16bf b, v8f c) {
  return __builtin_amdgcn_wmma_f32_16x16x32_bf16(false, a, false, b, (short)0, c,
                                                 false, false);
}

__device__ __forceinline__ v8f vzero8() {
  v8f z;
#pragma unroll
  for (int i = 0; i < 8; ++i) z[i] = 0.0f;
  return z;
}

// ---------------------------------------------------------------------------
// Routed GEMM:  C[idx[m], :] = A[idx[m], :] @ B_expert   (idx==nullptr => id)
// Block tile 128x128, K-step 32. LDS holds tiles in WMMA fragment order.
// ---------------------------------------------------------------------------
__global__ __launch_bounds__(256)
void gemm_routed_bf16(const float* __restrict__ A, int lda,
                      const float* __restrict__ B, int ldb, long bExpStride,
                      float* __restrict__ C, int ldc,
                      const int* __restrict__ idx, int chunkM, int K)
{
  __shared__ v16bf lds_a[8 * 32];
  __shared__ v16bf lds_b[8 * 32];

  const int tid   = threadIdx.x;
  const int s     = tid >> 5;          // subtile 0..7 (staging role)
  const int l     = tid & 31;
  const int rl    = l & 15;
  const int koff  = (l & 16) ? 8 : 0;  // lane-half K offset (WMMA bf16 layout)
  const int eBase = blockIdx.z * chunkM;
  const int mBase = blockIdx.y * 128;
  const int nBase = blockIdx.x * 128;

  const int sortedA = eBase + mBase + s * 16 + rl;
  const int arow    = idx ? idx[sortedA] : sortedA;
  const float* aptr = A + (size_t)arow * lda + koff;
  const float* bptr = B + (size_t)blockIdx.z * bExpStride +
                      (size_t)koff * ldb + (nBase + s * 16 + rl);

  const int wm = s >> 1;               // wave M index 0..3 (rows wm*32)
  const int wn = s & 1;                // wave N index 0..1 (cols wn*64)

  v8f acc[2][4];
#pragma unroll
  for (int i = 0; i < 2; ++i)
#pragma unroll
    for (int j = 0; j < 4; ++j) acc[i][j] = vzero8();

  for (int k0 = 0; k0 < K; k0 += 32) {
    // Prefetch next K-step (lowers to global_prefetch_b8).
    if (k0 + 32 < K) {
      __builtin_prefetch(aptr + k0 + 32, 0, 1);
      __builtin_prefetch(bptr + (size_t)(k0 + 32) * ldb, 0, 1);
    }
    lds_a[s * 32 + l] = mk_frag(aptr + k0, 1.0f);
    lds_b[s * 32 + l] = mk_frag_strided(bptr + (size_t)k0 * ldb, (size_t)ldb);
    __syncthreads();
    v16bf af0 = lds_a[(wm * 2 + 0) * 32 + l];
    v16bf af1 = lds_a[(wm * 2 + 1) * 32 + l];
#pragma unroll
    for (int j = 0; j < 4; ++j) {
      v16bf bf = lds_b[(wn * 4 + j) * 32 + l];
      acc[0][j] = wmma_bf16(af0, bf, acc[0][j]);
      acc[1][j] = wmma_bf16(af1, bf, acc[1][j]);
    }
    __syncthreads();
  }

  // epilogue: scatter 16x16 f32 C tiles (row = idx[sorted])
  const int rowhalf = (l & 16) ? 8 : 0;
#pragma unroll
  for (int i = 0; i < 2; ++i) {
    const int m0 = wm * 32 + i * 16;
    int orow[8];
#pragma unroll
    for (int r = 0; r < 8; ++r) {
      int sorted = eBase + mBase + m0 + rowhalf + r;
      orow[r] = idx ? idx[sorted] : sorted;
    }
#pragma unroll
    for (int j = 0; j < 4; ++j) {
      const int col = nBase + wn * 64 + j * 16 + rl;
#pragma unroll
      for (int r = 0; r < 8; ++r)
        C[(size_t)orow[r] * ldc + col] = acc[i][j][r];
    }
  }
}

// ---------------------------------------------------------------------------
// Fused RMSNorm (over HD=64) + NeoX RoPE, in place. One wave per (token,head).
// ---------------------------------------------------------------------------
__global__ __launch_bounds__(256)
void rmsnorm_rope(float* __restrict__ x, const float* __restrict__ w,
                  const int* __restrict__ positions, int nheads, int ld)
{
  const int pair = blockIdx.x * 8 + (threadIdx.x >> 5);
  const int lane = threadIdx.x & 31;
  const int tok  = pair / nheads;
  const int head = pair - tok * nheads;
  float* row = x + (size_t)tok * ld + head * HDIM;

  float a = row[lane];        // x1[lane]
  float b = row[lane + 32];   // x2[lane]
  float ss = a * a + b * b;
#pragma unroll
  for (int off = 16; off >= 1; off >>= 1) ss += __shfl_xor(ss, off, 32);
  const float inv = rsqrtf(ss * (1.0f / 64.0f) + 1e-6f);
  a = a * inv * w[lane];
  b = b * inv * w[lane + 32];

  const float pos  = (float)positions[tok];
  const float freq = pos * __powf(10000.0f, -(float)lane * (2.0f / 64.0f));
  float sn, cs;
  __sincosf(freq, &sn, &cs);
  row[lane]      = a * cs - b * sn;
  row[lane + 32] = b * cs + a * sn;
}

// ---------------------------------------------------------------------------
// Flash attention. grid = (NTOK/64, NHEADS), block = 128 (4 waves).
// Per 64-key block: the whole block cooperatively stages the shared K and V
// tiles into LDS in WMMA fragment order (4x less global traffic than
// per-wave loads), then each wave runs QK^T (8 WMMA), online softmax
// (16-lane shuffle reductions), P->LDS->A-frags, P.V (8 WMMA).
// ---------------------------------------------------------------------------
__global__ __launch_bounds__(128)
void attn_fa(const float* __restrict__ qb, const float* __restrict__ kb,
             const float* __restrict__ vb, float* __restrict__ ob)
{
  __shared__ v16bf lds_k[8][32];                    // [kf*4+nt][lane]
  __shared__ v16bf lds_v[8][32];                    // [kf*4+dt][lane]
  __shared__ __attribute__((aligned(16))) __bf16 lds_p[4][16][64];

  const int w       = threadIdx.x >> 5;
  const int lane    = threadIdx.x & 31;
  const int rl      = lane & 15;
  const int koff    = (lane & 16) ? 8 : 0;
  const int rowhalf = (lane & 16) ? 8 : 0;
  const int qblk    = blockIdx.x;
  const int h       = blockIdx.y;
  const int kvh     = h >> 2;                       // NH/NKV = 4
  const int qrow0   = qblk * 64 + w * 16;

  // Q fragments (16 x 64, two K-steps), pre-scaled by HD^-1/2.
  v16bf aq[2];
  {
    const float* qp = qb + (size_t)(qrow0 + rl) * (NHEADS * HDIM) + h * HDIM;
    aq[0] = mk_frag(qp + koff,      0.125f);
    aq[1] = mk_frag(qp + 32 + koff, 0.125f);
  }

  float m_run[8], l_run[8];
  v8f acc[4];
#pragma unroll
  for (int r = 0; r < 8; ++r) { m_run[r] = -3.0e38f; l_run[r] = 0.0f; }
#pragma unroll
  for (int d = 0; d < 4; ++d) acc[d] = vzero8();

  for (int kbk = 0; kbk <= qblk; ++kbk) {
    const int kbase = kbk * 64;

    // ---- cooperative staging: this wave stages fragment-sets 2w and 2w+1 ----
#pragma unroll
    for (int i = 0; i < 2; ++i) {
      const int f  = w * 2 + i;                     // 0..7
      const int kf = f >> 2;                        // K-step
      const int t  = f & 3;                         // nt (K) / dt (V)
      const int base = kf * 32 + koff;
      // K rows load directly as column-major B fragments
      const float* kp = kb + (size_t)(kbase + t * 16 + rl) * (NKVH * HDIM) + kvh * HDIM;
      lds_k[f][lane] = mk_frag(kp + base, 1.0f);
      // V columns as B fragments (strided gather, shared by all 4 waves)
      const float* vp = vb + (size_t)(kbase + base) * (NKVH * HDIM) + kvh * HDIM + t * 16 + rl;
      lds_v[f][lane] = mk_frag_strided(vp, (size_t)(NKVH * HDIM));
    }
    __syncthreads();

    // ---- S = Q . K^T ----
    v8f s[4];
#pragma unroll
    for (int nt = 0; nt < 4; ++nt) s[nt] = vzero8();
#pragma unroll
    for (int kf = 0; kf < 2; ++kf)
#pragma unroll
      for (int nt = 0; nt < 4; ++nt)
        s[nt] = wmma_bf16(aq[kf], lds_k[kf * 4 + nt][lane], s[nt]);

    // ---- causal mask (diagonal block only; positions are arange) ----
    if (kbk == qblk) {
#pragma unroll
      for (int nt = 0; nt < 4; ++nt) {
        const int j = kbase + nt * 16 + rl;
#pragma unroll
        for (int r = 0; r < 8; ++r)
          if (j > qrow0 + rowhalf + r) s[nt][r] = -1.0e30f;
      }
    }

    // ---- online softmax; each row lives in one half-wave's one VGPR ----
    float alpha[8];
#pragma unroll
    for (int r = 0; r < 8; ++r) {
      float mx = fmaxf(fmaxf(s[0][r], s[1][r]), fmaxf(s[2][r], s[3][r]));
#pragma unroll
      for (int off = 8; off >= 1; off >>= 1) mx = fmaxf(mx, __shfl_xor(mx, off, 32));
      const float mn = fmaxf(m_run[r], mx);
      alpha[r] = __expf(m_run[r] - mn);
      m_run[r] = mn;
      float rs = 0.0f;
#pragma unroll
      for (int nt = 0; nt < 4; ++nt) {
        const float p = __expf(s[nt][r] - mn);
        s[nt][r] = p;
        rs += p;
      }
#pragma unroll
      for (int off = 8; off >= 1; off >>= 1) rs += __shfl_xor(rs, off, 32);
      l_run[r] = l_run[r] * alpha[r] + rs;
    }

    // ---- P (C-layout) -> LDS row-major bf16; rescale O ----
#pragma unroll
    for (int nt = 0; nt < 4; ++nt)
#pragma unroll
      for (int r = 0; r < 8; ++r)
        lds_p[w][rowhalf + r][nt * 16 + rl] = f2bf(s[nt][r]);
#pragma unroll
    for (int d = 0; d < 4; ++d)
#pragma unroll
      for (int r = 0; r < 8; ++r) acc[d][r] *= alpha[r];

    // ---- O += P . V ----
#pragma unroll
    for (int kf = 0; kf < 2; ++kf) {
      const int base = kf * 32 + koff;
      v16bf ap;
#pragma unroll
      for (int j = 0; j < 8; ++j) {
        ap[j]     = lds_p[w][rl][base + j];
        ap[8 + j] = lds_p[w][rl][base + 16 + j];
      }
#pragma unroll
      for (int dt = 0; dt < 4; ++dt)
        acc[dt] = wmma_bf16(ap, lds_v[kf * 4 + dt][lane], acc[dt]);
    }
    __syncthreads();   // before next iteration overwrites lds_k / lds_v
  }

  // ---- epilogue: O /= l ----
#pragma unroll
  for (int dt = 0; dt < 4; ++dt)
#pragma unroll
    for (int r = 0; r < 8; ++r)
      ob[(size_t)(qrow0 + rowhalf + r) * (NHEADS * HDIM) + h * HDIM + dt * 16 + rl] =
          acc[dt][r] / l_run[r];
}

// ---------------------------------------------------------------------------
extern "C" void kernel_launch(void* const* d_in, const int* in_sizes, int n_in,
                              void* d_out, int out_size, void* d_ws, size_t ws_size,
                              hipStream_t stream)
{
  (void)in_sizes; (void)n_in; (void)out_size; (void)ws_size;
  const float* hidden    = (const float*)d_in[0];
  const int*   positions = (const int*)d_in[1];
  const int*   sort_idx  = (const int*)d_in[2];
  const float* q_proj_w  = (const float*)d_in[3];
  const float* o_proj_w  = (const float*)d_in[4];
  const float* k_w       = (const float*)d_in[5];
  const float* v_w       = (const float*)d_in[6];
  const float* q_norm_w  = (const float*)d_in[7];
  const float* k_norm_w  = (const float*)d_in[8];
  float* out = (float*)d_out;

  float* q = (float*)d_ws;                       // NTOK x 2048
  float* k = q + (size_t)NTOK * (NHEADS * HDIM); // NTOK x 512
  float* v = k + (size_t)NTOK * (NKVH * HDIM);   // NTOK x 512
  float* o = v + (size_t)NTOK * (NKVH * HDIM);   // NTOK x 2048

  const dim3 blk256(256);
  const int chunkM = NTOK / NEXP;                // 512

  // Q = routed(hidden @ q_proj_w[e])
  gemm_routed_bf16<<<dim3((NHEADS * HDIM) / 128, chunkM / 128, NEXP), blk256, 0, stream>>>(
      hidden, HIDDIM, q_proj_w, NHEADS * HDIM, (long)HIDDIM * NHEADS * HDIM,
      q, NHEADS * HDIM, sort_idx, chunkM, HIDDIM);
  // K = hidden @ k_w
  gemm_routed_bf16<<<dim3((NKVH * HDIM) / 128, NTOK / 128, 1), blk256, 0, stream>>>(
      hidden, HIDDIM, k_w, NKVH * HDIM, 0, k, NKVH * HDIM, nullptr, NTOK, HIDDIM);
  // V = hidden @ v_w
  gemm_routed_bf16<<<dim3((NKVH * HDIM) / 128, NTOK / 128, 1), blk256, 0, stream>>>(
      hidden, HIDDIM, v_w, NKVH * HDIM, 0, v, NKVH * HDIM, nullptr, NTOK, HIDDIM);

  // RMSNorm + RoPE (in place)
  rmsnorm_rope<<<dim3((NTOK * NHEADS) / 8), blk256, 0, stream>>>(
      q, q_norm_w, positions, NHEADS, NHEADS * HDIM);
  rmsnorm_rope<<<dim3((NTOK * NKVH) / 8), blk256, 0, stream>>>(
      k, k_norm_w, positions, NKVH, NKVH * HDIM);

  // Causal GQA flash attention
  attn_fa<<<dim3(NTOK / 64, NHEADS), dim3(128), 0, stream>>>(q, k, v, o);

  // out = routed(o @ o_proj_w[e])
  gemm_routed_bf16<<<dim3(HIDDIM / 128, chunkM / 128, NEXP), blk256, 0, stream>>>(
      o, NHEADS * HDIM, o_proj_w, HIDDIM, (long)(NHEADS * HDIM) * HIDDIM,
      out, HIDDIM, sort_idx, chunkM, NHEADS * HDIM);
}
